// opt_joint_map_all_86663850099029
// MI455X (gfx1250) — matrix operand order repaired and checked by reading, not compiled, not run on hardware
//
#include <hip/hip_runtime.h>

// MI455X / gfx1250, wave32. Matmuls via v_wmma_f32_16x16x32_f16.

typedef __attribute__((ext_vector_type(16))) _Float16 v16h;
typedef __attribute__((ext_vector_type(8)))  float    v8f;

constexpr int Bc = 512;   // batch
constexpr int Tc = 256;   // horizon
constexpr int Fc = 8;     // exo features
constexpr int Hc = 128;   // lstm hidden
constexpr int G4 = 512;   // 4*H gate width
constexpr int MHc = 512;  // mlp hidden
constexpr int BT = Bc * Tc;

#if __has_builtin(__builtin_amdgcn_exp2f)
#define FEXP2(x) __builtin_amdgcn_exp2f(x)
#else
#define FEXP2(x) exp2f(x)
#endif
#if __has_builtin(__builtin_amdgcn_rcpf)
#define FRCP(x) __builtin_amdgcn_rcpf(x)
#else
#define FRCP(x) (1.0f / (x))
#endif

// Branch-free activations: one v_exp_f32 + one v_rcp_f32 each.
__device__ __forceinline__ float sig_fast(float x) {
  return FRCP(1.0f + FEXP2(-1.4426950408889634f * x));
}
__device__ __forceinline__ float tanh_fast(float x) {
  // (e^2x - 1)/(e^2x + 1); saturates correctly at +/-1.
  return 1.0f - 2.0f * FRCP(1.0f + FEXP2(2.8853900817779268f * x));
}

// A-matrix (16x32 f16) lane layout: half-element e of lane (hi = lane/16):
//   e<8  -> K = hi*8 + e          (VGPR 0..3)
//   e>=8 -> K = 16 + hi*8 + (e-8) (VGPR 4..7)
__device__ __forceinline__ int a_k(int e, int hi) {
  return (e < 8) ? (hi * 8 + e) : (16 + hi * 8 + (e - 8));
}

__device__ __forceinline__ v8f wmma_f16(v16h a, v16h b, v8f c) {
  return __builtin_amdgcn_wmma_f32_16x16x32_f16(false, a, false, b, (short)0, c,
                                                false, false);
}

// ---------------------------------------------------------------------------
// LSTM kernel: grid (B/16, 3), 512 threads = 16 waves.
// Whh/Wih fragments register-resident across all T steps; h (f16), c (f32) in LDS.
// ---------------------------------------------------------------------------
__global__ __launch_bounds__(512) void lstm3_kernel(
    const float* __restrict__ X0,  const float* __restrict__ X1,  const float* __restrict__ X2,
    const float* __restrict__ Wih0, const float* __restrict__ Whh0, const float* __restrict__ bb0,
    const float* __restrict__ hw0,  const float* __restrict__ hb0,
    const float* __restrict__ Wih1, const float* __restrict__ Whh1, const float* __restrict__ bb1,
    const float* __restrict__ hw1,  const float* __restrict__ hb1,
    const float* __restrict__ Wih2, const float* __restrict__ Whh2, const float* __restrict__ bb2,
    const float* __restrict__ hw2,  const float* __restrict__ hb2,
    float* __restrict__ outBase)
{
  const int which = blockIdx.y;
  const float *X, *Wih, *Whh, *bias, *hw, *hbp;
  if (which == 0)      { X = X0; Wih = Wih0; Whh = Whh0; bias = bb0; hw = hw0; hbp = hb0; }
  else if (which == 1) { X = X1; Wih = Wih1; Whh = Whh1; bias = bb1; hw = hw1; hbp = hb1; }
  else                 { X = X2; Wih = Wih2; Whh = Whh2; bias = bb2; hw = hw2; hbp = hb2; }
  float* __restrict__ Dout = outBase + which * BT;

  const int tid  = threadIdx.x;
  const int wave = tid >> 5;
  const int lane = tid & 31;
  const int hi   = lane >> 4;
  const int ml   = lane & 15;
  const int b0   = blockIdx.x * 16;

  __shared__ _Float16 sH[16][Hc + 8];    // h_t (A-matrix source), f16
  __shared__ float    sC[16][Hc];        // c_t
  __shared__ float    sZ[16][G4 + 4];    // preactivations
  __shared__ float    sBias[G4];
  __shared__ float    sHw[Hc];
  __shared__ float    sPred[16];

  for (int i = tid; i < G4; i += 512) sBias[i] = bias[i];
  for (int i = tid; i < Hc; i += 512) sHw[i] = hw[i];
  for (int i = tid; i < 16 * Hc; i += 512) {
    int m = i / Hc, k = i % Hc;
    sH[m][k] = (_Float16)0.f;
    sC[m][k] = 0.f;
  }
  if (tid < 16) sPred[tid] = 0.f;

  // Resident weight fragments: 2 N-tiles/wave, 4 h-K-tiles + 1 padded x-tile.
  // B layout (32x16 f16): element e -> K = hi*16 + e, N = n0 + ml.
  v16h bh[2][4];
  v16h bx[2];
#pragma unroll
  for (int j = 0; j < 2; ++j) {
    const int n = (wave * 2 + j) * 16 + ml;
    const float* wsrc = Whh + n * Hc;
#pragma unroll
    for (int kt = 0; kt < 4; ++kt) {
      const float* s = wsrc + kt * 32 + hi * 16;
#pragma unroll
      for (int e = 0; e < 16; ++e) bh[j][kt][e] = (_Float16)s[e];
    }
#pragma unroll
    for (int e = 0; e < 16; ++e) {
      const int k = hi * 16 + e;
      bx[j][e] = (k < Fc) ? (_Float16)Wih[n * Fc + k] : (_Float16)0.f;
    }
  }
  const float hbv = hbp[0];
  __syncthreads();

  for (int t = 0; t < Tc; ++t) {
    // A fragment for exogenous x_t (K padded 8 -> 32 with zeros).
    v16h ax;
#pragma unroll
    for (int e = 0; e < 16; ++e) ax[e] = (_Float16)0.f;
    if (hi == 0) {
      const float* xp = X + ((size_t)(b0 + ml) * Tc + t) * Fc;
#pragma unroll
      for (int e = 0; e < 8; ++e) ax[e] = (_Float16)xp[e];
      __builtin_prefetch(xp + 8 * Fc, 0, 0);  // prefetch x for t+8
    }
    // A fragments for h (from LDS, f16).
    v16h ah[4];
#pragma unroll
    for (int kt = 0; kt < 4; ++kt) {
#pragma unroll
      for (int e = 0; e < 16; ++e) ah[kt][e] = sH[ml][kt * 32 + hi * 16 + e];
    }

#pragma unroll
    for (int j = 0; j < 2; ++j) {
      v8f acc;
#pragma unroll
      for (int r = 0; r < 8; ++r) acc[r] = 0.f;
      acc = wmma_f16(ax, bx[j], acc);
#pragma unroll
      for (int kt = 0; kt < 4; ++kt) acc = wmma_f16(ah[kt], bh[j][kt], acc);
      const int n0 = (wave * 2 + j) * 16;
#pragma unroll
      for (int r = 0; r < 8; ++r) sZ[hi * 8 + r][n0 + ml] = acc[r];  // M = 8*hi + r
    }
    __syncthreads();

    // Fused, branch-free gate math: 2048 (b,n') pairs over 512 threads.
    {
      const int bb = tid >> 5;
      const int nb = (tid & 31) * 4;
      float part = 0.f;
#pragma unroll
      for (int q = 0; q < 4; ++q) {
        const int np = nb + q;
        const float gi = sZ[bb][np]          + sBias[np];
        const float gf = sZ[bb][Hc + np]     + sBias[Hc + np];
        const float gg = sZ[bb][2 * Hc + np] + sBias[2 * Hc + np];
        const float go = sZ[bb][3 * Hc + np] + sBias[3 * Hc + np];
        const float c  = sig_fast(gf) * sC[bb][np] + sig_fast(gi) * tanh_fast(gg);
        const float h  = sig_fast(go) * tanh_fast(c);
        sC[bb][np] = c;
        sH[bb][np] = (_Float16)h;
        part += h * sHw[np];
      }
      atomicAdd(&sPred[bb], part);   // ds_add_f32
    }
    __syncthreads();
    if (tid < 16) {
      Dout[(size_t)(b0 + tid) * Tc + t] = sPred[tid] + hbv;
      sPred[tid] = 0.f;
    }
    __syncthreads();
  }
}

// ---------------------------------------------------------------------------
// W2 pre-transpose+convert: W2T[n][k] = (f16) W2[k][n]. 512 KB, L2-resident.
// ---------------------------------------------------------------------------
__global__ __launch_bounds__(256) void w2_transpose_f16(
    const float* __restrict__ W2, _Float16* __restrict__ W2T)
{
  const int i = blockIdx.x * 256 + threadIdx.x;    // 0 .. MHc*MHc-1
  const int n = i >> 9, k = i & (MHc - 1);
  W2T[(size_t)n * MHc + k] = (_Float16)W2[(size_t)k * MHc + n];
}

// ---------------------------------------------------------------------------
// MLP kernel: grid BT/32, 256 threads = 8 waves. M-tile 32, barrier-free K-loop.
// F16W=true: B fragments are 2x contiguous global_load_b128 from f16 W2T (L2).
// ---------------------------------------------------------------------------
template <bool F16W>
__global__ __launch_bounds__(256) void mlp_kernel(
    const float* __restrict__ Dbase,  // D1 | D2 | D3, each BT floats
    const float* __restrict__ W1, const float* __restrict__ b1,
    const void*  __restrict__ W2p, const float* __restrict__ b2,
    const float* __restrict__ W3, const float* __restrict__ b3,
    float* __restrict__ out4)
{
  const int tid  = threadIdx.x;
  const int wave = tid >> 5;
  const int lane = tid & 31;
  const int hi   = lane >> 4;
  const int ml   = lane & 15;
  const int m0   = blockIdx.x * 32;

  __shared__ _Float16 sH1[32][MHc + 8];  // layer-1 activations, f16 (~33 KB)
  __shared__ float    sOut[32];

  // Layer 1 (K=3, negligible FLOPs): 8 threads per row, 64 k's each.
  {
    const int m  = tid >> 3;
    const int k0 = (tid & 7) * 64;
    const int bt = m0 + m;
    const float x0 = Dbase[bt];
    const float x1 = Dbase[BT + bt];
    const float x2 = Dbase[2 * BT + bt];
#pragma unroll
    for (int q = 0; q < 64; ++q) {
      const int k = k0 + q;
      const float h = fmaxf(x0 * W1[k] + x1 * W1[MHc + k] + x2 * W1[2 * MHc + k] + b1[k], 0.f);
      sH1[m][k] = (_Float16)h;
    }
  }
  if (tid < 32) sOut[tid] = b3[0];
  __syncthreads();

  // Layer 2: 32x512 @ 512x512. 8 waves x 4 N-tiles x 2 M-tiles; no K barriers.
  v8f acc[2][4];
#pragma unroll
  for (int mt = 0; mt < 2; ++mt)
#pragma unroll
    for (int nt = 0; nt < 4; ++nt)
#pragma unroll
      for (int r = 0; r < 8; ++r) acc[mt][nt][r] = 0.f;

  const _Float16* W2h = (const _Float16*)W2p;  // [n][k] when F16W
  const float*    W2f = (const float*)W2p;     // [k][n] when !F16W

  for (int kc = 0; kc < MHc / 32; ++kc) {
    v16h a[2];
#pragma unroll
    for (int mt = 0; mt < 2; ++mt)
#pragma unroll
      for (int e = 0; e < 16; ++e)
        a[mt][e] = sH1[mt * 16 + ml][kc * 32 + hi * 16 + e];

#pragma unroll
    for (int nt = 0; nt < 4; ++nt) {
      const int n = (wave * 4 + nt) * 16 + ml;
      v16h bf;
      if (F16W) {
        const _Float16* bp = W2h + (size_t)n * MHc + kc * 32 + hi * 16;
#pragma unroll
        for (int e = 0; e < 16; ++e) bf[e] = bp[e];  // 2x b128 from L2
      } else {
#pragma unroll
        for (int e = 0; e < 16; ++e)
          bf[e] = (_Float16)W2f[(size_t)(kc * 32 + hi * 16 + e) * MHc + n];
      }
      acc[0][nt] = wmma_f16(a[0], bf, acc[0][nt]);
      acc[1][nt] = wmma_f16(a[1], bf, acc[1][nt]);
    }
  }

  // Epilogue: bias + ReLU + dot with W3, reduced per batch row via ds_add_f32.
  float part[2][8];
#pragma unroll
  for (int mt = 0; mt < 2; ++mt)
#pragma unroll
    for (int r = 0; r < 8; ++r) part[mt][r] = 0.f;
#pragma unroll
  for (int nt = 0; nt < 4; ++nt) {
    const int n = (wave * 4 + nt) * 16 + ml;
    const float w3 = W3[n];
    const float bn = b2[n];
#pragma unroll
    for (int mt = 0; mt < 2; ++mt)
#pragma unroll
      for (int r = 0; r < 8; ++r) {
        const float v = fmaxf(acc[mt][nt][r] + bn, 0.f);
        part[mt][r] += v * w3;
      }
  }
#pragma unroll
  for (int mt = 0; mt < 2; ++mt)
#pragma unroll
    for (int r = 0; r < 8; ++r)
      atomicAdd(&sOut[mt * 16 + hi * 8 + r], part[mt][r]);
  __syncthreads();
  if (tid < 32) out4[m0 + tid] = sOut[tid];
}

// ---------------------------------------------------------------------------
extern "C" void kernel_launch(void* const* d_in, const int* in_sizes, int n_in,
                              void* d_out, int out_size, void* d_ws, size_t ws_size,
                              hipStream_t stream) {
  (void)in_sizes; (void)n_in; (void)out_size;
  const float* Dis_out = (const float*)d_in[0];
  const float* Dis_rad = (const float*)d_in[1];
  const float* Dis_occ = (const float*)d_in[2];
  const float* l1_Wih = (const float*)d_in[3];
  const float* l1_Whh = (const float*)d_in[4];
  const float* l1_b   = (const float*)d_in[5];
  const float* l1_hw  = (const float*)d_in[6];
  const float* l1_hb  = (const float*)d_in[7];
  const float* l2_Wih = (const float*)d_in[8];
  const float* l2_Whh = (const float*)d_in[9];
  const float* l2_b   = (const float*)d_in[10];
  const float* l2_hw  = (const float*)d_in[11];
  const float* l2_hb  = (const float*)d_in[12];
  const float* l3_Wih = (const float*)d_in[13];
  const float* l3_Whh = (const float*)d_in[14];
  const float* l3_b   = (const float*)d_in[15];
  const float* l3_hw  = (const float*)d_in[16];
  const float* l3_hb  = (const float*)d_in[17];
  const float* W1 = (const float*)d_in[18];
  const float* b1 = (const float*)d_in[19];
  const float* W2 = (const float*)d_in[20];
  const float* b2 = (const float*)d_in[21];
  const float* W3 = (const float*)d_in[22];
  const float* b3 = (const float*)d_in[23];

  float* out = (float*)d_out;  // [D1 | D2 | D3 | c_p_F], each BT floats

  dim3 gL(Bc / 16, 3);
  lstm3_kernel<<<gL, 512, 0, stream>>>(
      Dis_out, Dis_rad, Dis_occ,
      l1_Wih, l1_Whh, l1_b, l1_hw, l1_hb,
      l2_Wih, l2_Whh, l2_b, l2_hw, l2_hb,
      l3_Wih, l3_Whh, l3_b, l3_hw, l3_hb,
      out);

  const size_t w2tBytes = (size_t)MHc * MHc * sizeof(_Float16);
  if (ws_size >= w2tBytes) {
    _Float16* W2T = (_Float16*)d_ws;
    w2_transpose_f16<<<(MHc * MHc) / 256, 256, 0, stream>>>(W2, W2T);
    mlp_kernel<true><<<BT / 32, 256, 0, stream>>>(out, W1, b1, (const void*)W2T,
                                                  b2, W3, b3, out + 3 * BT);
  } else {
    mlp_kernel<false><<<BT / 32, 256, 0, stream>>>(out, W1, b1, (const void*)W2,
                                                   b2, W3, b3, out + 3 * BT);
  }
}
